// DGCNNDense_60043642798277
// MI455X (gfx1250) — matrix operand order repaired and checked by reading, not compiled, not run on hardware
//
#include <hip/hip_runtime.h>

#define K_NN 30
#define NPTS 2048

typedef _Float16 h8   __attribute__((ext_vector_type(8)));
typedef _Float16 v16h __attribute__((ext_vector_type(16)));
typedef float    v8f  __attribute__((ext_vector_type(8)));
typedef float    v2f  __attribute__((ext_vector_type(2)));

__device__ __forceinline__ v16h combine16(h8 lo, h8 hi) {
  return __builtin_shufflevector(lo, hi, 0,1,2,3,4,5,6,7,8,9,10,11,12,13,14,15);
}

__device__ __forceinline__ v8f wmma_f16f32(v16h a, v16h b, v8f c) {
  // v_wmma_f32_16x16x32_f16: D = A(16x32 f16) * B(32x16 f16) + C(16x16 f32)
  return __builtin_amdgcn_wmma_f32_16x16x32_f16(false, a, false, b, (short)0, c, false, false);
}

__device__ __forceinline__ v8f wmma_f32x4(v2f a, v2f b, v8f c) {
  // v_wmma_f32_16x16x4_f32: exact f32 GEMM step on the matrix pipes
  return __builtin_amdgcn_wmma_f32_16x16x4_f32(false, a, false, b, (short)0, c, false, false);
}

// A fragment (f16, 16x32): lane half h holds K chunks [8h,8h+8) and [16+8h,16+8h+8).
__device__ __forceinline__ v16h load_a_frag(const _Float16* tile, int ldk, int m,
                                            int kbase, int half) {
  const _Float16* p = tile + (size_t)m * ldk + kbase + 8 * half;
  h8 lo = *(const h8*)p;
  h8 hi = *(const h8*)(p + 16);
  return combine16(lo, hi);
}

// B fragment (f16, 32x16): lane holds column n; lane half h holds K chunks [8h,8h+8) and [16+8h,16+8h+8).
__device__ __forceinline__ v16h load_b_frag(const _Float16* wT, int ldk, int n,
                                            int kbase, int half) {
  const _Float16* p = wT + (size_t)n * ldk + kbase + 8 * half;
  h8 lo = *(const h8*)p;
  h8 hi = *(const h8*)(p + 16);
  return combine16(lo, hi);
}

// ---------------------------------------------------------------------------
// Weight prep: dst[n*KP + k] = (k < K) ? (f16)src[k*N + n] : 0
// ---------------------------------------------------------------------------
__global__ void transpose_f16_kernel(const float* __restrict__ src,
                                     _Float16* __restrict__ dst,
                                     int Kdim, int Ndim, int KP) {
  int i = blockIdx.x * 256 + threadIdx.x;
  if (i >= Ndim * KP) return;
  int n = i / KP, k = i % KP;
  dst[(size_t)n * KP + k] = (k < Kdim) ? (_Float16)src[(size_t)k * Ndim + n]
                                       : (_Float16)0.0f;
}

// ---------------------------------------------------------------------------
// kNN (small C, scalar distances): one wave per query point.
// ---------------------------------------------------------------------------
template <int C>
__global__ __launch_bounds__(128) void knn_kernel(const float* __restrict__ x,
                                                  int* __restrict__ idx,
                                                  int B, int N) {
  __shared__ float dist[4][NPTS];   // 32 KB
  int wave = threadIdx.x >> 5;
  int lane = threadIdx.x & 31;
  int q = blockIdx.x * 4 + wave;
  int b = q / N, n = q % N;
  const float* xb = x + (size_t)b * N * C;

  float qf[C];
#pragma unroll
  for (int c = 0; c < C; ++c) qf[c] = xb[(size_t)n * C + c];

  for (int m = lane; m < N; m += 32) {
    const float* xm = xb + (size_t)m * C;
    float d = 0.0f;
#pragma unroll
    for (int c = 0; c < C; ++c) { float t = qf[c] - xm[c]; d += t * t; }
    dist[wave][m] = d;
  }
  __syncthreads();

  int* out = idx + (size_t)q * 32;
  for (int iter = 0; iter < K_NN; ++iter) {
    float best = 3.4e38f;
    int bi = 0;
    for (int m = lane; m < N; m += 32) {
      float d = dist[wave][m];
      if (d < best) { best = d; bi = m; }
    }
#pragma unroll
    for (int off = 16; off > 0; off >>= 1) {
      float ob = __shfl_xor(best, off, 32);
      int   oi = __shfl_xor(bi, off, 32);
      if (ob < best || (ob == best && oi < bi)) { best = ob; bi = oi; }
    }
    if (lane == 0) { out[iter] = bi; dist[wave][bi] = 3.4e38f; }
    __syncthreads();
  }
}

// ---------------------------------------------------------------------------
// kNN (C=64) via f32 WMMA gram matrix. One block = 16 query rows, 8 waves.
// rank value r[j] = |x_j|^2 - 2*(x_i . x_j)  (same ordering as squared dist).
// Uses the 320KB CDNA5 LDS: 16x2048 f32 distance tile resident in LDS.
// ---------------------------------------------------------------------------
__global__ __launch_bounds__(256) void knn_gram_kernel(const float* __restrict__ x,
                                                       int* __restrict__ idx,
                                                       int B, int N) {
  __shared__ __align__(16) float Q[16 * 64];      // 4 KB query tile
  __shared__ float norms[NPTS];                   // 8 KB
  __shared__ float dist[16 * NPTS];               // 128 KB
  int wave = threadIdx.x >> 5;
  int lane = threadIdx.x & 31;
  int tid = threadIdx.x;
  int qt = blockIdx.x;                 // 16-query tile index (N % 16 == 0)
  int b  = (qt * 16) / N;
  int n0 = (qt * 16) % N;
  const float* xb = x + (size_t)b * N * 64;

  // Stage the query tile and compute candidate norms cooperatively.
  for (int i = tid; i < 16 * 64; i += 256)
    Q[i] = xb[(size_t)(n0 + (i >> 6)) * 64 + (i & 63)];
  for (int m = tid; m < N; m += 256) {
    const float* xm = xb + (size_t)m * 64;
    float s = 0.0f;
#pragma unroll 8
    for (int c = 0; c < 64; ++c) s += xm[c] * xm[c];
    norms[m] = s;
  }
  __syncthreads();

  int half = lane >> 4, col = lane & 15;

  // Hoist f32 A fragments: 16 K-steps of 4; lane half h holds K = 2h, 2h+1.
  v2f af[16];
#pragma unroll
  for (int kt = 0; kt < 16; ++kt)
    af[kt] = *(const v2f*)&Q[col * 64 + kt * 4 + 2 * half];

  // Each wave owns 256 candidate columns -> 16 N-tiles of 16.
  for (int t = 0; t < 16; ++t) {
    int cbase = wave * 256 + t * 16;
    if (t + 1 < 16)   // prefetch next candidate block (global_prefetch_b8)
      __builtin_prefetch(xb + (size_t)(cbase + 16 + col) * 64, 0, 3);
    v8f acc = (v8f){0,0,0,0,0,0,0,0};
#pragma unroll
    for (int kt = 0; kt < 16; ++kt) {
      v2f bf = *(const v2f*)(xb + (size_t)(cbase + col) * 64 + kt * 4 + 2 * half);
      acc = wmma_f32x4(af[kt], bf, acc);
    }
    int cand = cbase + col;
    float nj = norms[cand];
#pragma unroll
    for (int r = 0; r < 8; ++r)
      dist[(size_t)(r + 8 * half) * NPTS + cand] = nj - 2.0f * acc[r];
  }
  __syncthreads();

  // Selection: each wave owns 2 rows; 30 iterative min-selects per row.
  for (int rr = 0; rr < 2; ++rr) {
    int row = wave * 2 + rr;
    int q = qt * 16 + row;
    int* out = idx + (size_t)q * 32;
    float* drow = dist + (size_t)row * NPTS;
    for (int iter = 0; iter < K_NN; ++iter) {
      float best = 3.4e38f;
      int bi = 0;
      for (int m = lane; m < NPTS; m += 32) {
        float d = drow[m];
        if (d < best) { best = d; bi = m; }
      }
#pragma unroll
      for (int off = 16; off > 0; off >>= 1) {
        float ob = __shfl_xor(best, off, 32);
        int   oi = __shfl_xor(bi, off, 32);
        if (ob < best || (ob == best && oi < bi)) { best = ob; bi = oi; }
      }
      if (lane == 0) { out[iter] = bi; drow[bi] = 3.4e38f; }
      __syncthreads();
    }
  }
}

// ---------------------------------------------------------------------------
// EdgeConv: one wave per point. A = 32x2C edge features (rows 30/31 zero),
// layer0: [32,2C]@[2C,64] (WMMA), relu; layer1: [32,64]@[64,64]; row-max.
// ---------------------------------------------------------------------------
template <int C>
__global__ __launch_bounds__(128) void edge_conv_kernel(
    const float* __restrict__ x, const int* __restrict__ idx,
    const _Float16* __restrict__ w0T, const float* __restrict__ b0,
    const _Float16* __restrict__ w1T, const float* __restrict__ b1,
    float* __restrict__ out, int B, int N) {
  constexpr int KP = (2 * C + 31) & ~31;            // 32 (C=3) or 128 (C=64)
  __shared__ __align__(16) _Float16 Atile[4][32 * KP];
  __shared__ __align__(16) _Float16 Htile[4][32 * 64];

  int wave = threadIdx.x >> 5;
  int lane = threadIdx.x & 31;
  int q = blockIdx.x * 4 + wave;
  int b = q / N, n = q % N;
  const float* xb = x + (size_t)b * N * C;
  _Float16* A = Atile[wave];

  // Build edge-feature tile: row e = [x_i, x_j - x_i], pad K and rows to 32.
  {
    int e = lane;
    _Float16* row = A + (size_t)e * KP;
    if (e < K_NN) {
      int j = idx[(size_t)q * 32 + e];
      const float* xi = xb + (size_t)n * C;
      const float* xj = xb + (size_t)j * C;
#pragma unroll
      for (int c = 0; c < C; ++c) {
        float vi = xi[c];
        row[c]     = (_Float16)vi;
        row[C + c] = (_Float16)(xj[c] - vi);
      }
#pragma unroll
      for (int c = 2 * C; c < KP; ++c) row[c] = (_Float16)0.0f;
    } else {
#pragma unroll
      for (int c = 0; c < KP; ++c) row[c] = (_Float16)0.0f;
    }
  }
  __syncthreads();

  int half = lane >> 4, col = lane & 15;

  // ---- layer 0 ----
  v8f acc[2][4];
#pragma unroll
  for (int mt = 0; mt < 2; ++mt)
#pragma unroll
    for (int nt = 0; nt < 4; ++nt) acc[mt][nt] = (v8f){0,0,0,0,0,0,0,0};

#pragma unroll
  for (int kt = 0; kt < KP / 32; ++kt) {
    v16h a0 = load_a_frag(A, KP, col, kt * 32, half);
    v16h a1 = load_a_frag(A, KP, 16 + col, kt * 32, half);
#pragma unroll
    for (int nt = 0; nt < 4; ++nt) {
      v16h bf = load_b_frag(w0T, KP, nt * 16 + col, kt * 32, half);
      acc[0][nt] = wmma_f16f32(a0, bf, acc[0][nt]);
      acc[1][nt] = wmma_f16f32(a1, bf, acc[1][nt]);
    }
  }

  // bias + relu, stage h1 as f16 [32][64]
  _Float16* H = Htile[wave];
#pragma unroll
  for (int nt = 0; nt < 4; ++nt) {
    float bb = b0[nt * 16 + col];
#pragma unroll
    for (int mt = 0; mt < 2; ++mt) {
#pragma unroll
      for (int r = 0; r < 8; ++r) {
        float v = acc[mt][nt][r] + bb;
        v = v > 0.0f ? v : 0.0f;
        H[(size_t)(mt * 16 + r + 8 * half) * 64 + nt * 16 + col] = (_Float16)v;
      }
    }
  }
  __syncthreads();

  // ---- layer 1 ----
  v8f acc2[2][4];
#pragma unroll
  for (int mt = 0; mt < 2; ++mt)
#pragma unroll
    for (int nt = 0; nt < 4; ++nt) acc2[mt][nt] = (v8f){0,0,0,0,0,0,0,0};

#pragma unroll
  for (int kt = 0; kt < 2; ++kt) {
    v16h a0 = load_a_frag(H, 64, col, kt * 32, half);
    v16h a1 = load_a_frag(H, 64, 16 + col, kt * 32, half);
#pragma unroll
    for (int nt = 0; nt < 4; ++nt) {
      v16h bf = load_b_frag(w1T, 64, nt * 16 + col, kt * 32, half);
      acc2[0][nt] = wmma_f16f32(a0, bf, acc2[0][nt]);
      acc2[1][nt] = wmma_f16f32(a1, bf, acc2[1][nt]);
    }
  }

  // max over the 30 valid edge rows (exclude padded rows 30,31), + bias
  float res[4];
#pragma unroll
  for (int nt = 0; nt < 4; ++nt) {
    float v = -3.4e38f;
#pragma unroll
    for (int r = 0; r < 8; ++r) v = fmaxf(v, acc2[0][nt][r]);   // rows 0..15
#pragma unroll
    for (int r = 0; r < 8; ++r) {                                // rows 16..31
      if (half == 0 || r < 6) v = fmaxf(v, acc2[1][nt][r]);      // M<30 only
    }
    v = fmaxf(v, __shfl_xor(v, 16, 32));   // fold lane halves (same column)
    res[nt] = v;
  }
  if (lane < 16) {
    float* o = out + (size_t)q * 64;
#pragma unroll
    for (int nt = 0; nt < 4; ++nt)
      o[nt * 16 + lane] = res[nt] + b1[nt * 16 + lane];
  }
}

// ---------------------------------------------------------------------------
// Final MLP: one wave per 16 points. 192 -> 1024 -> 256 -> 128 (WMMA) -> 3.
// ---------------------------------------------------------------------------
__global__ __launch_bounds__(32) void final_mlp_kernel(
    const float* __restrict__ x1, const float* __restrict__ x2,
    const float* __restrict__ x3,
    const _Float16* __restrict__ w0T, const float* __restrict__ b0,
    const _Float16* __restrict__ w1T, const float* __restrict__ b1,
    const _Float16* __restrict__ w2T, const float* __restrict__ b2,
    const float* __restrict__ finw, const float* __restrict__ finb,
    float* __restrict__ out, int B, int N) {
  __shared__ __align__(16) _Float16 Ain[16 * 192];   // 6 KB
  __shared__ __align__(16) _Float16 H1[16 * 1024];   // 32 KB
  __shared__ __align__(16) _Float16 H2[16 * 256];    // 8 KB
  __shared__ float H3[16 * 128];                     // 8 KB

  int lane = threadIdx.x & 31;
  int p0 = blockIdx.x * 16;
  int half = lane >> 4, col = lane & 15;

  // Fill Ain[16][192] = concat(x1, x2, x3) as f16; 2 lanes per point row.
  {
    int row = lane & 15, seg = lane >> 4;   // seg: features [0,96) or [96,192)
    int p = p0 + row;
    _Float16* dst = Ain + (size_t)row * 192;
    const float* s1 = x1 + (size_t)p * 64;
    const float* s2 = x2 + (size_t)p * 64;
    const float* s3 = x3 + (size_t)p * 64;
    for (int f = 96 * seg; f < 96 * seg + 96; ++f) {
      float v = (f < 64) ? s1[f] : ((f < 128) ? s2[f - 64] : s3[f - 128]);
      dst[f] = (_Float16)v;
    }
  }
  __syncthreads();

  // ---- layer 1: [16,192] @ [192,1024], relu ----
  v16h afr[6];
#pragma unroll
  for (int kt = 0; kt < 6; ++kt) afr[kt] = load_a_frag(Ain, 192, col, kt * 32, half);

  for (int nt = 0; nt < 64; ++nt) {
    v8f acc = (v8f){0,0,0,0,0,0,0,0};
#pragma unroll
    for (int kt = 0; kt < 6; ++kt) {
      v16h bf = load_b_frag(w0T, 192, nt * 16 + col, kt * 32, half);
      acc = wmma_f16f32(afr[kt], bf, acc);
    }
    float bb = b0[nt * 16 + col];
#pragma unroll
    for (int r = 0; r < 8; ++r) {
      float v = acc[r] + bb;
      v = v > 0.0f ? v : 0.0f;
      H1[(size_t)(r + 8 * half) * 1024 + nt * 16 + col] = (_Float16)v;
    }
  }
  __syncthreads();

  // ---- layer 2: [16,1024] @ [1024,256], relu ----
  for (int nt = 0; nt < 16; ++nt) {
    v8f acc = (v8f){0,0,0,0,0,0,0,0};
    for (int kt = 0; kt < 32; ++kt) {
      v16h af = load_a_frag(H1, 1024, col, kt * 32, half);
      v16h bf = load_b_frag(w1T, 1024, nt * 16 + col, kt * 32, half);
      acc = wmma_f16f32(af, bf, acc);
    }
    float bb = b1[nt * 16 + col];
#pragma unroll
    for (int r = 0; r < 8; ++r) {
      float v = acc[r] + bb;
      v = v > 0.0f ? v : 0.0f;
      H2[(size_t)(r + 8 * half) * 256 + nt * 16 + col] = (_Float16)v;
    }
  }
  __syncthreads();

  // ---- layer 3: [16,256] @ [256,128], no relu, keep f32 ----
  for (int nt = 0; nt < 8; ++nt) {
    v8f acc = (v8f){0,0,0,0,0,0,0,0};
#pragma unroll
    for (int kt = 0; kt < 8; ++kt) {
      v16h af = load_a_frag(H2, 256, col, kt * 32, half);
      v16h bf = load_b_frag(w2T, 256, nt * 16 + col, kt * 32, half);
      acc = wmma_f16f32(af, bf, acc);
    }
    float bb = b2[nt * 16 + col];
#pragma unroll
    for (int r = 0; r < 8; ++r)
      H3[(size_t)(r + 8 * half) * 128 + nt * 16 + col] = acc[r] + bb;
  }
  __syncthreads();

  // ---- final 128 -> 3 (scalar; tiny) ----
  for (int o = lane; o < 48; o += 32) {
    int m = o & 15, j = o >> 4;           // j in 0..2
    const float* h = H3 + (size_t)m * 128;
    float s = finb[j];
    for (int c = 0; c < 128; ++c) s += h[c] * finw[c * 3 + j];
    out[(size_t)(p0 + m) * 3 + j] = s;
  }
}

// ---------------------------------------------------------------------------
// Launcher
// ---------------------------------------------------------------------------
extern "C" void kernel_launch(void* const* d_in, const int* in_sizes, int n_in,
                              void* d_out, int out_size, void* d_ws, size_t ws_size,
                              hipStream_t stream) {
  const float* pos    = (const float*)d_in[0];
  const float* c1_w0  = (const float*)d_in[1];
  const float* c1_b0  = (const float*)d_in[2];
  const float* c1_w1  = (const float*)d_in[3];
  const float* c1_b1  = (const float*)d_in[4];
  const float* c2_w0  = (const float*)d_in[5];
  const float* c2_b0  = (const float*)d_in[6];
  const float* c2_w1  = (const float*)d_in[7];
  const float* c2_b1  = (const float*)d_in[8];
  const float* c3_w0  = (const float*)d_in[9];
  const float* c3_b0  = (const float*)d_in[10];
  const float* c3_w1  = (const float*)d_in[11];
  const float* c3_b1  = (const float*)d_in[12];
  const float* mlp_w0 = (const float*)d_in[13];
  const float* mlp_b0 = (const float*)d_in[14];
  const float* mlp_w1 = (const float*)d_in[15];
  const float* mlp_b1 = (const float*)d_in[16];
  const float* mlp_w2 = (const float*)d_in[17];
  const float* mlp_b2 = (const float*)d_in[18];
  const float* fin_w  = (const float*)d_in[19];
  const float* fin_b  = (const float*)d_in[20];
  (void)n_in; (void)out_size; (void)ws_size;

  const int N = NPTS;
  const int B = in_sizes[0] / (N * 3);
  const int P = B * N;

  size_t off = 0;
  auto alloc = [&](size_t bytes) -> void* {
    void* p = (char*)d_ws + off;
    off = (off + bytes + 255) & ~(size_t)255;
    return p;
  };
  float* x1 = (float*)alloc((size_t)P * 64 * sizeof(float));
  float* x2 = (float*)alloc((size_t)P * 64 * sizeof(float));
  float* x3 = (float*)alloc((size_t)P * 64 * sizeof(float));
  int* idxb = (int*)alloc((size_t)P * 32 * sizeof(int));
  _Float16* w0_1T = (_Float16*)alloc(64 * 32 * 2);
  _Float16* w1_1T = (_Float16*)alloc(64 * 64 * 2);
  _Float16* w0_2T = (_Float16*)alloc(64 * 128 * 2);
  _Float16* w1_2T = (_Float16*)alloc(64 * 64 * 2);
  _Float16* w0_3T = (_Float16*)alloc(64 * 128 * 2);
  _Float16* w1_3T = (_Float16*)alloc(64 * 64 * 2);
  _Float16* mw0T  = (_Float16*)alloc((size_t)1024 * 192 * 2);
  _Float16* mw1T  = (_Float16*)alloc((size_t)256 * 1024 * 2);
  _Float16* mw2T  = (_Float16*)alloc((size_t)128 * 256 * 2);

  auto T = [&](const float* src, _Float16* dst, int Kd, int Nd, int KP) {
    int total = Nd * KP;
    transpose_f16_kernel<<<(total + 255) / 256, 256, 0, stream>>>(src, dst, Kd, Nd, KP);
  };
  T(c1_w0, w0_1T, 6, 64, 32);
  T(c1_w1, w1_1T, 64, 64, 64);
  T(c2_w0, w0_2T, 128, 64, 128);
  T(c2_w1, w1_2T, 64, 64, 64);
  T(c3_w0, w0_3T, 128, 64, 128);
  T(c3_w1, w1_3T, 64, 64, 64);
  T(mlp_w0, mw0T, 192, 1024, 192);
  T(mlp_w1, mw1T, 1024, 256, 1024);
  T(mlp_w2, mw2T, 256, 128, 256);

  // EdgeConv 1 (C=3): scalar kNN (trivial FLOPs)
  knn_kernel<3><<<P / 4, 128, 0, stream>>>(pos, idxb, B, N);
  edge_conv_kernel<3><<<P / 4, 128, 0, stream>>>(pos, idxb, w0_1T, c1_b0,
                                                 w1_1T, c1_b1, x1, B, N);
  // EdgeConv 2 (C=64): WMMA gram kNN
  knn_gram_kernel<<<P / 16, 256, 0, stream>>>(x1, idxb, B, N);
  edge_conv_kernel<64><<<P / 4, 128, 0, stream>>>(x1, idxb, w0_2T, c2_b0,
                                                  w1_2T, c2_b1, x2, B, N);
  // EdgeConv 3 (C=64): WMMA gram kNN
  knn_gram_kernel<<<P / 16, 256, 0, stream>>>(x2, idxb, B, N);
  edge_conv_kernel<64><<<P / 4, 128, 0, stream>>>(x2, idxb, w0_3T, c3_b0,
                                                  w1_3T, c3_b1, x3, B, N);
  // Final per-point MLP
  final_mlp_kernel<<<P / 16, 32, 0, stream>>>(x1, x2, x3, mw0T, mlp_b0,
                                              mw1T, mlp_b1, mw2T, mlp_b2,
                                              fin_w, fin_b, (float*)d_out, B, N);
}
